// Experts_46540265619726
// MI455X (gfx1250) — matrix operand order repaired and checked by reading, not compiled, not run on hardware
//
#include <hip/hip_runtime.h>
#include <hip/hip_bf16.h>

typedef __attribute__((ext_vector_type(16))) _Float16 v16h;
typedef __attribute__((ext_vector_type(8)))  float    v8f;

// Problem constants (match reference)
#define NE 8
#define NT 2048
#define ND 2048
#define NH 4096

// Tiling (kernel 1)
#define BM 128   // token tile
#define BN 64    // output-column tile (kernel 1)
#define BK 32    // k tile (== WMMA K for f16)
#define LDA 40   // padded LDS k-stride (halfs) for A tiles
#define LDB 40   // padded LDS k-stride (halfs) for B tiles (rows are n)
// Kernel 2 widens N
#define BN2 128

union Frag16 {
    v16h v;
    uint4 q[2];
};

__device__ __forceinline__ v8f wmma_f16(const v16h& a, const v16h& b, const v8f& c) {
    return __builtin_amdgcn_wmma_f32_16x16x32_f16(
        /*neg_a=*/false, a, /*neg_b=*/false, b,
        /*c_mod=*/(short)0, c, /*reuse_a=*/false, /*reuse_b=*/false);
}

// ---------------------------------------------------------------------------
// Kernel 1: h[e,t,n] = silu(x@gate_w) * (x@inner_w), h stored as f16 in ws.
// Double-buffered LDS, one barrier per k-step, 8 WMMA/step/wave.
// Grid: (NH/BN, NT/BM, NE), block: 256 threads (8 waves).
// ---------------------------------------------------------------------------
__global__ __launch_bounds__(256, 1)
void experts_gate_inner_kernel(const float* __restrict__ x,
                               const float* __restrict__ gate_w,
                               const float* __restrict__ inner_w,
                               _Float16* __restrict__ h) {
    const int e  = blockIdx.z;
    const int tb = blockIdx.y * BM;   // token base
    const int nb = blockIdx.x * BN;   // H-column base

    const float* xe = x       + (size_t)e * NT * ND;
    const float* gw = gate_w  + (size_t)e * ND * NH;
    const float* iw = inner_w + (size_t)e * ND * NH;
    _Float16*    he = h       + (size_t)e * NT * NH;

    __shared__ _Float16 As[2][BM * LDA];
    __shared__ _Float16 Bg[2][BN * LDB];
    __shared__ _Float16 Bi[2][BN * LDB];

    const int tid  = threadIdx.x;
    const int lane = tid & 31;
    const int wave = tid >> 5;
    const int wm   = wave & 3;    // 4 waves along M
    const int wn   = wave >> 2;   // 2 waves along N
    const int half = lane >> 4;
    const int lm   = lane & 15;

    int aRow[4], aCol[4];
#pragma unroll
    for (int it = 0; it < 4; ++it) {
        const int idx = tid + it * 256;   // 0..1023 float4 slots
        aRow[it] = idx >> 3;              // 8 float4 per 32-col row
        aCol[it] = (idx & 7) << 2;
    }
    int bK[2], bC[2];
#pragma unroll
    for (int it = 0; it < 2; ++it) {
        const int idx = tid + it * 256;   // 0..511 float4 slots
        bK[it] = idx >> 4;                // 16 float4 per 64-col row
        bC[it] = (idx & 15) << 2;
    }

    float4 aReg[4], gReg[2], iReg[2];

    auto g_load = [&](int kb) {
#pragma unroll
        for (int it = 0; it < 4; ++it)
            aReg[it] = *reinterpret_cast<const float4*>(
                xe + (size_t)(tb + aRow[it]) * ND + kb + aCol[it]);
#pragma unroll
        for (int it = 0; it < 2; ++it) {
            const size_t go = (size_t)(kb + bK[it]) * NH + nb + bC[it];
            gReg[it] = *reinterpret_cast<const float4*>(gw + go);
            iReg[it] = *reinterpret_cast<const float4*>(iw + go);
        }
    };
    auto l_store = [&](int buf) {
#pragma unroll
        for (int it = 0; it < 4; ++it) {
            _Float16* dst = &As[buf][aRow[it] * LDA + aCol[it]];
            dst[0] = (_Float16)aReg[it].x; dst[1] = (_Float16)aReg[it].y;
            dst[2] = (_Float16)aReg[it].z; dst[3] = (_Float16)aReg[it].w;
        }
#pragma unroll
        for (int it = 0; it < 2; ++it) {
            Bg[buf][(bC[it] + 0) * LDB + bK[it]] = (_Float16)gReg[it].x;
            Bg[buf][(bC[it] + 1) * LDB + bK[it]] = (_Float16)gReg[it].y;
            Bg[buf][(bC[it] + 2) * LDB + bK[it]] = (_Float16)gReg[it].z;
            Bg[buf][(bC[it] + 3) * LDB + bK[it]] = (_Float16)gReg[it].w;
            Bi[buf][(bC[it] + 0) * LDB + bK[it]] = (_Float16)iReg[it].x;
            Bi[buf][(bC[it] + 1) * LDB + bK[it]] = (_Float16)iReg[it].y;
            Bi[buf][(bC[it] + 2) * LDB + bK[it]] = (_Float16)iReg[it].z;
            Bi[buf][(bC[it] + 3) * LDB + bK[it]] = (_Float16)iReg[it].w;
        }
    };

    v8f accg[2][2], acci[2][2];
#pragma unroll
    for (int i = 0; i < 2; ++i)
#pragma unroll
        for (int j = 0; j < 2; ++j) { accg[i][j] = (v8f)0.0f; acci[i][j] = (v8f)0.0f; }

    g_load(0);
    l_store(0);
    __syncthreads();

    const int nsteps = ND / BK;
    for (int s = 0; s < nsteps; ++s) {
        const int cur = s & 1;
        if (s + 1 < nsteps) g_load((s + 1) * BK);  // overlap with compute below

        Frag16 a[2];
#pragma unroll
        for (int mt = 0; mt < 2; ++mt) {
            const _Float16* src = &As[cur][(wm * 32 + mt * 16 + lm) * LDA];
            a[mt].q[0] = *reinterpret_cast<const uint4*>(src + 8 * half);
            a[mt].q[1] = *reinterpret_cast<const uint4*>(src + 16 + 8 * half);
        }
        Frag16 fbg[2], fbi[2];
#pragma unroll
        for (int nt = 0; nt < 2; ++nt) {
            const _Float16* sg = &Bg[cur][(wn * 32 + nt * 16 + lm) * LDB + 8 * half];
            fbg[nt].q[0] = *reinterpret_cast<const uint4*>(sg);
            fbg[nt].q[1] = *reinterpret_cast<const uint4*>(sg + 16);
            const _Float16* si = &Bi[cur][(wn * 32 + nt * 16 + lm) * LDB + 8 * half];
            fbi[nt].q[0] = *reinterpret_cast<const uint4*>(si);
            fbi[nt].q[1] = *reinterpret_cast<const uint4*>(si + 16);
        }
#pragma unroll
        for (int mt = 0; mt < 2; ++mt)
#pragma unroll
            for (int nt = 0; nt < 2; ++nt) {
                accg[mt][nt] = wmma_f16(a[mt].v, fbg[nt].v, accg[mt][nt]);
                acci[mt][nt] = wmma_f16(a[mt].v, fbi[nt].v, acci[mt][nt]);
            }

        if (s + 1 < nsteps) l_store(cur ^ 1);      // write the other buffer
        __syncthreads();
    }

    // ---- epilogue: h = silu(g) * i, store f16 (approx rcp is fine for f16) ----
#pragma unroll
    for (int mt = 0; mt < 2; ++mt)
#pragma unroll
        for (int nt = 0; nt < 2; ++nt) {
            const int mbase = tb + wm * 32 + mt * 16 + 8 * half;
            const int n     = nb + wn * 32 + nt * 16 + lm;
#pragma unroll
            for (int i = 0; i < 8; ++i) {
                const float g  = accg[mt][nt][i];
                const float sg = __builtin_amdgcn_rcpf(1.0f + __expf(-g));
                const float hv = g * sg * acci[mt][nt][i];
                he[(size_t)(mbase + i) * NH + n] = (_Float16)hv;
            }
        }
}

// ---------------------------------------------------------------------------
// Kernel 2: out[e,t,d] = h(f16) @ out_w (f32 -> f16), f32 accumulate/output.
// BN2=128: each wave owns 32x64 (2x4 frags) -> 8 WMMA per k-step.
// Grid: (ND/BN2, NT/BM, NE), block: 256 threads.
// ---------------------------------------------------------------------------
__global__ __launch_bounds__(256, 1)
void experts_out_kernel(const _Float16* __restrict__ h,
                        const float* __restrict__ out_w,
                        float* __restrict__ out) {
    const int e  = blockIdx.z;
    const int tb = blockIdx.y * BM;
    const int nb = blockIdx.x * BN2;   // D-column base

    const _Float16* he = h     + (size_t)e * NT * NH;
    const float*    ow = out_w + (size_t)e * NH * ND;
    float*          oe = out   + (size_t)e * NT * ND;

    __shared__ _Float16 As[2][BM * LDA];
    __shared__ _Float16 Bs[2][BN2 * LDB];

    const int tid  = threadIdx.x;
    const int lane = tid & 31;
    const int wave = tid >> 5;
    const int wm   = wave & 3;    // 4 waves along M (32 rows each)
    const int wn   = wave >> 2;   // 2 waves along N (64 cols each)
    const int half = lane >> 4;
    const int lm   = lane & 15;

    int aRow[2], aCol[2];
#pragma unroll
    for (int it = 0; it < 2; ++it) {
        const int idx = tid + it * 256;   // 0..511 uint4 (8-half) slots
        aRow[it] = idx >> 2;              // 4 slots per 32-half row
        aCol[it] = (idx & 3) << 3;
    }
    int bK[4], bC[4];
#pragma unroll
    for (int it = 0; it < 4; ++it) {
        const int idx = tid + it * 256;   // 0..1023 float4 slots (32k x 128n)
        bK[it] = idx >> 5;                // 32 float4 per 128-col row
        bC[it] = (idx & 31) << 2;
    }

    uint4  aReg[2];
    float4 bReg[4];

    auto g_load = [&](int kb) {
#pragma unroll
        for (int it = 0; it < 2; ++it)
            aReg[it] = *reinterpret_cast<const uint4*>(
                he + (size_t)(tb + aRow[it]) * NH + kb + aCol[it]);
#pragma unroll
        for (int it = 0; it < 4; ++it)
            bReg[it] = *reinterpret_cast<const float4*>(
                ow + (size_t)(kb + bK[it]) * ND + nb + bC[it]);
    };
    auto l_store = [&](int buf) {
#pragma unroll
        for (int it = 0; it < 2; ++it)
            *reinterpret_cast<uint4*>(&As[buf][aRow[it] * LDA + aCol[it]]) = aReg[it];
#pragma unroll
        for (int it = 0; it < 4; ++it) {
            Bs[buf][(bC[it] + 0) * LDB + bK[it]] = (_Float16)bReg[it].x;
            Bs[buf][(bC[it] + 1) * LDB + bK[it]] = (_Float16)bReg[it].y;
            Bs[buf][(bC[it] + 2) * LDB + bK[it]] = (_Float16)bReg[it].z;
            Bs[buf][(bC[it] + 3) * LDB + bK[it]] = (_Float16)bReg[it].w;
        }
    };

    v8f acc[2][4];
#pragma unroll
    for (int i = 0; i < 2; ++i)
#pragma unroll
        for (int j = 0; j < 4; ++j) acc[i][j] = (v8f)0.0f;

    g_load(0);
    l_store(0);
    __syncthreads();

    const int nsteps = NH / BK;
    for (int s = 0; s < nsteps; ++s) {
        const int cur = s & 1;
        if (s + 1 < nsteps) g_load((s + 1) * BK);

        Frag16 a[2];
#pragma unroll
        for (int mt = 0; mt < 2; ++mt) {
            const _Float16* src = &As[cur][(wm * 32 + mt * 16 + lm) * LDA];
            a[mt].q[0] = *reinterpret_cast<const uint4*>(src + 8 * half);
            a[mt].q[1] = *reinterpret_cast<const uint4*>(src + 16 + 8 * half);
        }
        Frag16 b[4];
#pragma unroll
        for (int nt = 0; nt < 4; ++nt) {
            const _Float16* sb = &Bs[cur][(wn * 64 + nt * 16 + lm) * LDB + 8 * half];
            b[nt].q[0] = *reinterpret_cast<const uint4*>(sb);
            b[nt].q[1] = *reinterpret_cast<const uint4*>(sb + 16);
        }
#pragma unroll
        for (int mt = 0; mt < 2; ++mt)
#pragma unroll
            for (int nt = 0; nt < 4; ++nt)
                acc[mt][nt] = wmma_f16(a[mt].v, b[nt].v, acc[mt][nt]);

        if (s + 1 < nsteps) l_store(cur ^ 1);
        __syncthreads();
    }

#pragma unroll
    for (int mt = 0; mt < 2; ++mt)
#pragma unroll
        for (int nt = 0; nt < 4; ++nt) {
            const int mbase = tb + wm * 32 + mt * 16 + 8 * half;
            const int n     = nb + wn * 64 + nt * 16 + lm;
#pragma unroll
            for (int i = 0; i < 8; ++i)
                oe[(size_t)(mbase + i) * ND + n] = acc[mt][nt][i];
        }
}

extern "C" void kernel_launch(void* const* d_in, const int* in_sizes, int n_in,
                              void* d_out, int out_size, void* d_ws, size_t ws_size,
                              hipStream_t stream) {
    (void)in_sizes; (void)n_in; (void)out_size; (void)ws_size;

    const float* x       = (const float*)d_in[0];
    // d_in[1] = num_tokens_per_expert (int32) — unused by the reference math.
    const float* gate_w  = (const float*)d_in[2];
    const float* inner_w = (const float*)d_in[3];
    const float* out_w   = (const float*)d_in[4];
    float*       out     = (float*)d_out;

    _Float16* hbuf = (_Float16*)d_ws;   // needs E*T*H*2 = 128 MiB

    dim3 block(256, 1, 1);
    dim3 grid1(NH / BN,  NT / BM, NE);   // (64, 16, 8)
    dim3 grid2(ND / BN2, NT / BM, NE);   // (16, 16, 8)

    experts_gate_inner_kernel<<<grid1, block, 0, stream>>>(x, gate_w, inner_w, hbuf);
    experts_out_kernel<<<grid2, block, 0, stream>>>(hbuf, out_w, out);
}
